// SparseDynamicWindowAttention_57836029608076
// MI455X (gfx1250) — compile-verified, matrix operand order checked by reading, not run on hardware
//
#include <hip/hip_runtime.h>
#include <hip/hip_bf16.h>
#include <math.h>

typedef _Float16 f16;
typedef __attribute__((ext_vector_type(16))) _Float16 v16h;
typedef __attribute__((ext_vector_type(8)))  float    v8f;

#define WINSZ  16
#define DIMC   256
#define NHEADS 8
#define BATCH  2
#define HH     256
#define WW2    256
#define KSEL   64              // 16*16*0.25
#define NWIN   (BATCH*KSEL)    // 128

// ---------------- workspace layout (all offsets multiple of 512B) ----------
#define SUMS_OFF   ((size_t)0)
#define SUMS_BYTES ((size_t)BATCH*HH*WW2*DIMC*4)           // 134,217,728
#define CNTS_OFF   (SUMS_OFF + SUMS_BYTES)
#define CNTS_BYTES ((size_t)BATCH*HH*WW2*4)                // 524,288
#define SD1_OFF    (CNTS_OFF + CNTS_BYTES)
#define SD1_BYTES  ((size_t)BATCH*4*HH*WW2*4)              // 2,097,152
#define SD2_OFF    (SD1_OFF + SD1_BYTES)
#define SD2_BYTES  ((size_t)BATCH*4*256*4)                 // 8,192
#define SD3_OFF    (SD2_OFF + SD2_BYTES)
#define SD3_BYTES  SD2_BYTES
#define WINH_OFF   (SD3_OFF + SD3_BYTES)
#define WINW_OFF   (WINH_OFF + 512)
#define WINHD_OFF  (WINW_OFF + 512)
#define PE_OFF     (WINHD_OFF + 512)
#define PE_BYTES   ((size_t)NHEADS*256*256*4)              // 2,097,152
#define XB32_OFF   (PE_OFF + PE_BYTES)
#define XB32_BYTES ((size_t)NWIN*256*DIMC*4)               // 33,554,432
#define XB16_OFF   (XB32_OFF + XB32_BYTES)
#define XB16_BYTES ((size_t)NWIN*256*DIMC*2)               // 16,777,216
#define INW16_OFF  (XB16_OFF + XB16_BYTES)
#define INW16_BYTES ((size_t)768*256*2)
#define OUTW16_OFF (INW16_OFF + INW16_BYTES)
#define OUTW16_BYTES ((size_t)256*256*2)
#define QKV16_OFF  (OUTW16_OFF + OUTW16_BYTES)
#define QKV16_BYTES ((size_t)NWIN*256*768*2)               // 50,331,648
#define O16_OFF    (QKV16_OFF + QKV16_BYTES)
#define O16_BYTES  ((size_t)NWIN*256*DIMC*2)               // 16,777,216

// ---------------- helpers --------------------------------------------------
__device__ __forceinline__ float sigm(float x){ return 1.f/(1.f+__expf(-x)); }
__device__ __forceinline__ float silu(float x){ return x*sigm(x); }

// Load a 16-wide K fragment (A-row or B-col) per the CDNA5 16-bit layout:
// element i in [0,8): K = kb+i ; element i in [8,16): K = kb+16+(i-8).
// Caller passes p = row/col base + kb, where kb = (lane>>4)*8.
__device__ __forceinline__ v16h frag_ld(const f16* p){
  v16h f;
#pragma unroll
  for (int i=0;i<8;++i){ f[i] = p[i]; f[i+8] = p[i+16]; }
  return f;
}

__device__ __forceinline__ v8f wmma32(v16h a, v16h b, v8f c){
  return __builtin_amdgcn_wmma_f32_16x16x32_f16(false, a, false, b, (short)0, c, false, false);
}

// ---------------- kernels --------------------------------------------------
__global__ void k_zero(float* p, long long n){
  long long i = (long long)blockIdx.x*blockDim.x + threadIdx.x;
  long long s = (long long)gridDim.x*blockDim.x;
  for (; i<n; i+=s) p[i] = 0.f;
}

// silu(conv3x3 C=256 -> 4), x is NHWC. One block per (b,h), one thread per w.
__global__ void __launch_bounds__(256) k_conv1(const float* __restrict__ x,
    const float* __restrict__ w1, const float* __restrict__ b1, float* __restrict__ sd1){
  __shared__ float wl[9*256*4];                 // [k9][c][oc]
  for (int idx=threadIdx.x; idx<9*256*4; idx+=256){
    int oc = idx & 3, c = (idx>>2)&255, k9 = idx>>10;
    wl[idx] = w1[((oc*256 + c)*3 + k9/3)*3 + (k9%3)];
  }
  __syncthreads();
  int b = blockIdx.x >> 8, h = blockIdx.x & 255, w = threadIdx.x;
  float a0=b1[0], a1=b1[1], a2=b1[2], a3=b1[3];
  for (int ky=0;ky<3;++ky){
    int y = h+ky-1; if ((unsigned)y >= 256u) continue;
    for (int kx=0;kx<3;++kx){
      int xw = w+kx-1; if ((unsigned)xw >= 256u) continue;
      const float* xp = x + ((size_t)(b*256+y)*256 + xw)*256;
      const float* wp = wl + (ky*3+kx)*1024;
      for (int c=0;c<256;++c){
        float xv = xp[c];
        a0 += xv*wp[c*4+0]; a1 += xv*wp[c*4+1];
        a2 += xv*wp[c*4+2]; a3 += xv*wp[c*4+3];
      }
    }
  }
  size_t base = ((size_t)(b*4)*256 + h)*256 + w;
  sd1[base            ] = silu(a0);
  sd1[base +   65536  ] = silu(a1);
  sd1[base + 2*65536  ] = silu(a2);
  sd1[base + 3*65536  ] = silu(a3);
}

// silu(conv 16x16 stride16 4->4). One block per (b,oc), thread per (oy,ox).
__global__ void __launch_bounds__(256) k_conv2(const float* __restrict__ sd1,
    const float* __restrict__ w2, const float* __restrict__ b2, float* __restrict__ sd2){
  int b = blockIdx.x >> 2, oc = blockIdx.x & 3;
  int oy = threadIdx.x >> 4, ox = threadIdx.x & 15;
  float acc = b2[oc];
  for (int ic=0; ic<4; ++ic){
    const float* ip = sd1 + ((size_t)(b*4+ic)*256 + oy*16)*256 + ox*16;
    const float* wp = w2 + ((oc*4+ic)*256);
    for (int ky=0;ky<16;++ky)
      for (int kx=0;kx<16;++kx)
        acc += ip[ky*256+kx]*wp[ky*16+kx];
  }
  sd2[(size_t)blockIdx.x*256 + threadIdx.x] = silu(acc);
}

// conv3x3 4->4 pad1 on 16x16 grid (raw, no activation)
__global__ void __launch_bounds__(256) k_conv3(const float* __restrict__ sd2,
    const float* __restrict__ w3, const float* __restrict__ b3, float* __restrict__ sd3){
  int b = blockIdx.x >> 2, oc = blockIdx.x & 3;
  int oy = threadIdx.x >> 4, ox = threadIdx.x & 15;
  float acc = b3[oc];
  for (int ic=0; ic<4; ++ic){
    const float* ip = sd2 + (size_t)(b*4+ic)*256;
    const float* wp = w3 + (oc*4+ic)*9;
    for (int ky=0;ky<3;++ky){
      int y = oy+ky-1; if ((unsigned)y >= 16u) continue;
      for (int kx=0;kx<3;++kx){
        int xx = ox+kx-1; if ((unsigned)xx >= 16u) continue;
        acc += ip[y*16+xx]*wp[ky*3+kx];
      }
    }
  }
  sd3[(size_t)blockIdx.x*256 + threadIdx.x] = acc;
}

// top-64 per batch by rank (softmax is monotone -> rank on raw scores).
__global__ void __launch_bounds__(256) k_topk(const float* __restrict__ sd3,
    int* winh, int* winw, float* winhard){
  int b = blockIdx.x, t = threadIdx.x;
  __shared__ float s[256];
  s[t] = sd3[(size_t)(b*4)*256 + t];
  __syncthreads();
  float st = s[t];
  int rank = 0;
  for (int j=0;j<256;++j){ float sj = s[j]; rank += (sj > st) || (sj == st && j < t); }
  if (rank < KSEL){
    int slot = b*KSEL + rank;
    float g    = sigm(sd3[(size_t)(b*4+1)*256 + t]);
    float offh = sigm(sd3[(size_t)(b*4+2)*256 + t]);
    float offw = sigm(sd3[(size_t)(b*4+3)*256 + t]);
    int h_off = (int)(offh*16.f - 8.f);   // trunc toward zero, matches astype(int32)
    int w_off = (int)(offw*16.f - 8.f);
    int hidx = t>>4, widx = t&15;
    int h1 = min(max(hidx*16 + h_off, 0), 240);
    int w1 = min(max(widx*16 + w_off, 0), 240);
    winh[slot] = h1; winw[slot] = w1;
    winhard[slot] = (g > 0.5f) ? 1.f : 0.f;
  }
}

// expand relative-position table to pe_full[h][q][k]
__global__ void __launch_bounds__(256) k_pe(const float* __restrict__ pet, float* __restrict__ pef){
  int idx = blockIdx.x*blockDim.x + threadIdx.x;   // 8*256*256 exactly
  int h = idx >> 16, q = (idx>>8)&255, k = idx & 255;
  int ry = (q>>4) - (k>>4) + 15;
  int rx = (q&15) - (k&15) + 15;
  pef[idx] = pet[(ry*31+rx)*NHEADS + h];
}

// bilinear window gather -> xb32 (f32, kept for delta) and xb16 (f16 GEMM A)
__global__ void __launch_bounds__(256) k_gather(const float* __restrict__ x,
    const int* __restrict__ winh, const int* __restrict__ winw,
    float* __restrict__ xb32, f16* __restrict__ xb16){
  int n = blockIdx.x, p = threadIdx.x;
  int b = n >> 6;
  int h1 = winh[n], w1 = winw[n];
  int py = p>>4, px = p&15;
  float yy = fminf((float)(h1+py)+0.5f, 255.f);
  float xf = fminf((float)(w1+px)+0.5f, 255.f);
  int y0 = (int)yy; float ly = yy - (float)y0; int y1 = min(y0+1, 255);
  int x0 = (int)xf; float lx = xf - (float)x0; int x1 = min(x0+1, 255);
  float w00=(1.f-ly)*(1.f-lx), w01=(1.f-ly)*lx, w10=ly*(1.f-lx), w11=ly*lx;
  const float* p00 = x + ((size_t)(b*256+y0)*256 + x0)*256;
  const float* p01 = x + ((size_t)(b*256+y0)*256 + x1)*256;
  const float* p10 = x + ((size_t)(b*256+y1)*256 + x0)*256;
  const float* p11 = x + ((size_t)(b*256+y1)*256 + x1)*256;
  float* o32 = xb32 + ((size_t)n*256 + p)*256;
  f16*   o16 = xb16 + ((size_t)n*256 + p)*256;
  for (int c=0;c<256;c+=4){
    float4 a = *(const float4*)(p00+c);
    float4 bq= *(const float4*)(p01+c);
    float4 cq= *(const float4*)(p10+c);
    float4 d = *(const float4*)(p11+c);
    float4 v;
    v.x = w00*a.x + w01*bq.x + w10*cq.x + w11*d.x;
    v.y = w00*a.y + w01*bq.y + w10*cq.y + w11*d.y;
    v.z = w00*a.z + w01*bq.z + w10*cq.z + w11*d.z;
    v.w = w00*a.w + w01*bq.w + w10*cq.w + w11*d.w;
    *(float4*)(o32+c) = v;
    o16[c]=(f16)v.x; o16[c+1]=(f16)v.y; o16[c+2]=(f16)v.z; o16[c+3]=(f16)v.w;
  }
}

__global__ void __launch_bounds__(256) k_cvtw(const float* __restrict__ inw,
    const float* __restrict__ outw, f16* __restrict__ inw16, f16* __restrict__ outw16){
  int i = blockIdx.x*blockDim.x + threadIdx.x;    // 768*256 threads
  if (i < 768*256)  inw16[i]  = (f16)inw[i];
  if (i < 256*256)  outw16[i] = (f16)outw[i];
}

// QKV GEMM: per block one (window, M-tile); 8 waves x 6 N-tiles = 48 of 768 cols.
__global__ void __launch_bounds__(256) k_qkv(const f16* __restrict__ xb16,
    const f16* __restrict__ w, const float* __restrict__ bias, f16* __restrict__ qkv){
  __shared__ f16 al[16*264];                       // 16 x 256 (+8 pad)
  int n = blockIdx.x >> 4, mt = blockIdx.x & 15;
  int t = threadIdx.x;
  {
    int row = t>>4, c0 = (t&15)*16;
    const f16* src = xb16 + ((size_t)(n*256 + mt*16 + row)*256) + c0;
    f16* dst = al + row*264 + c0;
#pragma unroll
    for (int i=0;i<16;++i) dst[i] = src[i];
  }
  __syncthreads();
  int wave = t>>5, lane = t&31, l16 = lane&15, kb = (lane>>4)<<3;
  v8f acc[6];
#pragma unroll
  for (int j=0;j<6;++j){
#pragma unroll
    for (int r=0;r<8;++r) acc[j][r] = 0.f;
  }
#pragma unroll
  for (int ks=0; ks<8; ++ks){
    int k0 = ks*32;
    v16h af = frag_ld(al + l16*264 + k0 + kb);
#pragma unroll
    for (int j=0;j<6;++j){
      int f0 = (wave*6+j)*16;
      v16h bf = frag_ld(w + (size_t)(f0 + l16)*256 + k0 + kb);
      acc[j] = wmma32(af, bf, acc[j]);
    }
  }
#pragma unroll
  for (int j=0;j<6;++j){
    int col = (wave*6+j)*16 + l16;
    float bz = bias[col];
#pragma unroll
    for (int r=0;r<8;++r){
      int tok = mt*16 + r + kb;                    // kb doubles as D-row offset (0/8)
      qkv[((size_t)n*256 + tok)*768 + col] = (f16)(acc[j][r] + bz);
    }
  }
}

// Attention: one block per (window, head). K and V^T staged in LDS; P staged
// per-wave in LDS scratch to reshape C-layout -> A-fragment layout.
__global__ void __launch_bounds__(256) k_attn(const f16* __restrict__ qkv,
    const float* __restrict__ pe, f16* __restrict__ o16){
  __shared__ f16 kl[256*40];       // k rows (tok, d) stride 40
  __shared__ f16 vt[32*264];       // v^T rows (d, tok) stride 264
  __shared__ f16 ps[8*16*40];      // per-wave P scratch 16x32 (+8 pad)
  int n = blockIdx.x >> 3, h = blockIdx.x & 7;
  int t = threadIdx.x;
  {
    const f16* kp = qkv + ((size_t)n*256 + t)*768 + 256 + h*32;
    const f16* vp = qkv + ((size_t)n*256 + t)*768 + 512 + h*32;
#pragma unroll
    for (int d=0; d<32; ++d){ kl[t*40 + d] = kp[d]; vt[d*264 + t] = vp[d]; }
  }
  __syncthreads();
  int wave = t>>5, lane = t&31, l16 = lane&15, kb = (lane>>4)<<3;
  const float scale = 0.17677669529663687f;        // 1/sqrt(32)
  f16* pw = ps + wave*(16*40);
  for (int mi=0; mi<2; ++mi){
    int mt = wave + mi*8;
    v16h qf = frag_ld(qkv + ((size_t)n*256 + mt*16 + l16)*768 + h*32 + kb);
    v8f st[16];
#pragma unroll
    for (int kt=0; kt<16; ++kt){
      v16h kf = frag_ld(kl + (kt*16 + l16)*40 + kb);
      v8f z;
#pragma unroll
      for (int r=0;r<8;++r) z[r] = 0.f;
      v8f s = wmma32(qf, kf, z);
#pragma unroll
      for (int r=0;r<8;++r){
        int qrow = mt*16 + r + kb;
        int kcol = kt*16 + l16;
        st[kt][r] = s[r]*scale + pe[((size_t)h*256 + qrow)*256 + kcol];
      }
    }
    // row softmax: each row lives in one 16-lane half across 16 tiles
#pragma unroll
    for (int r=0;r<8;++r){
      float m = -1e30f;
#pragma unroll
      for (int kt=0;kt<16;++kt) m = fmaxf(m, st[kt][r]);
      m = fmaxf(m, __shfl_xor(m, 1, 32));
      m = fmaxf(m, __shfl_xor(m, 2, 32));
      m = fmaxf(m, __shfl_xor(m, 4, 32));
      m = fmaxf(m, __shfl_xor(m, 8, 32));
      float sum = 0.f;
#pragma unroll
      for (int kt=0;kt<16;++kt){ float e = __expf(st[kt][r]-m); st[kt][r] = e; sum += e; }
      sum += __shfl_xor(sum, 1, 32);
      sum += __shfl_xor(sum, 2, 32);
      sum += __shfl_xor(sum, 4, 32);
      sum += __shfl_xor(sum, 8, 32);
      float inv = 1.f/sum;
#pragma unroll
      for (int kt=0;kt<16;++kt) st[kt][r] *= inv;
    }
    // O = P @ V : pair S tiles into 16x32 A fragments via LDS scratch
    v8f oa[2];
#pragma unroll
    for (int nv=0;nv<2;++nv){
#pragma unroll
      for (int r=0;r<8;++r) oa[nv][r] = 0.f;
    }
    for (int kp2=0; kp2<8; ++kp2){
#pragma unroll
      for (int r=0;r<8;++r){
        int row = r + kb;
        pw[row*40 + l16]      = (f16)st[2*kp2  ][r];
        pw[row*40 + 16 + l16] = (f16)st[2*kp2+1][r];
      }
      asm volatile("" ::: "memory");   // same-wave LDS is in-order; stop compiler reordering
      v16h pf = frag_ld(pw + l16*40 + kb);
#pragma unroll
      for (int nv=0; nv<2; ++nv){
        v16h vf = frag_ld(vt + (nv*16 + l16)*264 + kp2*32 + kb);
        oa[nv] = wmma32(pf, vf, oa[nv]);
      }
      asm volatile("" ::: "memory");
    }
#pragma unroll
    for (int nv=0;nv<2;++nv){
#pragma unroll
      for (int r=0;r<8;++r){
        int tok = mt*16 + r + kb;
        o16[((size_t)n*256 + tok)*256 + h*32 + nv*16 + l16] = (f16)oa[nv][r];
      }
    }
  }
}

// Out projection + delta scatter. Blocks for hard==0 windows exit immediately.
__global__ void __launch_bounds__(256) k_oproj(const f16* __restrict__ o16,
    const f16* __restrict__ w, const float* __restrict__ bias,
    const float* __restrict__ xb32, const int* __restrict__ winh,
    const int* __restrict__ winw, const float* __restrict__ winhard,
    float* __restrict__ sums){
  int n = blockIdx.x >> 4, mt = blockIdx.x & 15;
  if (winhard[n] == 0.f) return;                   // uniform per block
  __shared__ f16 al[16*264];
  int t = threadIdx.x;
  {
    int row = t>>4, c0 = (t&15)*16;
    const f16* src = o16 + ((size_t)(n*256 + mt*16 + row)*256) + c0;
    f16* dst = al + row*264 + c0;
#pragma unroll
    for (int i=0;i<16;++i) dst[i] = src[i];
  }
  __syncthreads();
  int wave = t>>5, lane = t&31, l16 = lane&15, kb = (lane>>4)<<3;
  v8f acc[2];
#pragma unroll
  for (int j=0;j<2;++j){
#pragma unroll
    for (int r=0;r<8;++r) acc[j][r] = 0.f;
  }
#pragma unroll
  for (int ks=0; ks<8; ++ks){
    int k0 = ks*32;
    v16h af = frag_ld(al + l16*264 + k0 + kb);
#pragma unroll
    for (int j=0;j<2;++j){
      int f0 = (wave*2+j)*16;
      v16h bf = frag_ld(w + (size_t)(f0 + l16)*256 + k0 + kb);
      acc[j] = wmma32(af, bf, acc[j]);
    }
  }
  int h1 = winh[n], w1 = winw[n], b = n>>6;
#pragma unroll
  for (int j=0;j<2;++j){
    int col = (wave*2+j)*16 + l16;
    float bz = bias[col];
#pragma unroll
    for (int r=0;r<8;++r){
      int tok = mt*16 + r + kb;
      float xa = acc[j][r] + bz;
      float d  = xa - xb32[((size_t)n*256 + tok)*256 + col];
      int pix = (b*256 + h1 + (tok>>4))*256 + (w1 + (tok&15));
      atomicAdd(sums + (size_t)pix*256 + col, d);
    }
  }
}

__global__ void __launch_bounds__(256) k_cnts(const int* __restrict__ winh,
    const int* __restrict__ winw, const float* __restrict__ winhard, float* __restrict__ cnts){
  int n = blockIdx.x, p = threadIdx.x;
  if (winhard[n] == 0.f) return;
  int b = n>>6;
  int pix = (b*256 + winh[n] + (p>>4))*256 + (winw[n] + (p&15));
  atomicAdd(cnts + pix, 1.f);
}

// Fused: xin = x + mean(sums/cnts), depthwise 3x3 conv, NHWC out
__global__ void __launch_bounds__(256) k_final(const float* __restrict__ x,
    const float* __restrict__ sums, const float* __restrict__ cnts,
    const float* __restrict__ ocw, const float* __restrict__ ocb, float* __restrict__ out){
  int blk = blockIdx.x;
  int b = blk >> 16, h = (blk>>8)&255, w = blk & 255;
  int c = threadIdx.x;
  float acc = ocb[c];
#pragma unroll
  for (int ky=0;ky<3;++ky){
    int y = h+ky-1; if ((unsigned)y >= 256u) continue;
#pragma unroll
    for (int kx=0;kx<3;++kx){
      int xw = w+kx-1; if ((unsigned)xw >= 256u) continue;
      size_t pix = (size_t)(b*256+y)*256 + xw;
      float cn = cnts[pix];
      float xin = x[pix*256 + c];
      if (cn > 0.f) xin += sums[pix*256 + c] / fmaxf(cn, 1.f);
      acc += ocw[c*9 + ky*3 + kx]*xin;
    }
  }
  size_t opix = (size_t)(b*256+h)*256 + w;
  __builtin_prefetch(x + (opix+1)*256 + c, 0, 1);  // global_prefetch_b8
  out[opix*256 + c] = acc;
}

// ---------------- launch ---------------------------------------------------
extern "C" void kernel_launch(void* const* d_in, const int* in_sizes, int n_in,
                              void* d_out, int out_size, void* d_ws, size_t ws_size,
                              hipStream_t stream){
  (void)in_sizes; (void)n_in; (void)out_size; (void)ws_size;
  const float* x     = (const float*)d_in[0];
  const float* sd_w1 = (const float*)d_in[1];
  const float* sd_b1 = (const float*)d_in[2];
  const float* sd_w2 = (const float*)d_in[3];
  const float* sd_b2 = (const float*)d_in[4];
  const float* sd_w3 = (const float*)d_in[5];
  const float* sd_b3 = (const float*)d_in[6];
  const float* in_w  = (const float*)d_in[7];
  const float* in_b  = (const float*)d_in[8];
  const float* pe_t  = (const float*)d_in[9];
  const float* out_w = (const float*)d_in[10];
  const float* out_b = (const float*)d_in[11];
  const float* oc_w  = (const float*)d_in[12];
  const float* oc_b  = (const float*)d_in[13];

  char* ws = (char*)d_ws;
  float* sums   = (float*)(ws + SUMS_OFF);
  float* cnts   = (float*)(ws + CNTS_OFF);
  float* sd1    = (float*)(ws + SD1_OFF);
  float* sd2    = (float*)(ws + SD2_OFF);
  float* sd3    = (float*)(ws + SD3_OFF);
  int*   winh   = (int*)  (ws + WINH_OFF);
  int*   winw   = (int*)  (ws + WINW_OFF);
  float* winhd  = (float*)(ws + WINHD_OFF);
  float* pef    = (float*)(ws + PE_OFF);
  float* xb32   = (float*)(ws + XB32_OFF);
  f16*   xb16   = (f16*)  (ws + XB16_OFF);
  f16*   inw16  = (f16*)  (ws + INW16_OFF);
  f16*   outw16 = (f16*)  (ws + OUTW16_OFF);
  f16*   qkv16  = (f16*)  (ws + QKV16_OFF);
  f16*   o16    = (f16*)  (ws + O16_OFF);

  // sums + cnts are contiguous: zero in one pass
  long long nz = (long long)BATCH*HH*WW2*DIMC + (long long)BATCH*HH*WW2;
  k_zero  <<<4096, 256, 0, stream>>>(sums, nz);
  k_conv1 <<<BATCH*256, 256, 0, stream>>>(x, sd_w1, sd_b1, sd1);
  k_conv2 <<<BATCH*4,   256, 0, stream>>>(sd1, sd_w2, sd_b2, sd2);
  k_conv3 <<<BATCH*4,   256, 0, stream>>>(sd2, sd_w3, sd_b3, sd3);
  k_topk  <<<BATCH,     256, 0, stream>>>(sd3, winh, winw, winhd);
  k_pe    <<<2048,      256, 0, stream>>>(pe_t, pef);
  k_gather<<<NWIN,      256, 0, stream>>>(x, winh, winw, xb32, xb16);
  k_cvtw  <<<768,       256, 0, stream>>>(in_w, out_w, inw16, outw16);
  k_qkv   <<<NWIN*16,   256, 0, stream>>>(xb16, inw16, in_b, qkv16);
  k_attn  <<<NWIN*NHEADS,256,0, stream>>>(qkv16, pef, o16);
  k_oproj <<<NWIN*16,   256, 0, stream>>>(o16, outw16, out_b, xb32, winh, winw, winhd, sums);
  k_cnts  <<<NWIN,      256, 0, stream>>>(winh, winw, winhd, cnts);
  k_final <<<BATCH*HH*WW2, 256, 0, stream>>>(x, sums, cnts, oc_w, oc_b, (float*)d_out);
}
